// MLMCrossMultiHeadAttention_55602646614530
// MI455X (gfx1250) — matrix operand order repaired and checked by reading, not compiled, not run on hardware
//
#include <hip/hip_runtime.h>
#include <hip/hip_bf16.h>
#include <math.h>

// ---------------------------------------------------------------------------
// MI455X (gfx1250, wave32) implementation.
// All GEMMs are NT (out[m,n] = dot(Arow_m, Brow_n), K contiguous in both),
// done with v_wmma_f32_16x16x32_bf16: f32 loads -> bf16 fragments -> f32 acc.
// ---------------------------------------------------------------------------

typedef __attribute__((ext_vector_type(16))) __bf16 v16bf;
typedef __attribute__((ext_vector_type(8)))  float  v8f;

#define D_DIM 2048
#define N1    250
#define N2    27

// ---- fragment packing -----------------------------------------------------

__device__ __forceinline__ v16bf pack16(float4 a, float4 b, float4 c, float4 d) {
  v16bf r;
  r[0]  = (__bf16)a.x; r[1]  = (__bf16)a.y; r[2]  = (__bf16)a.z; r[3]  = (__bf16)a.w;
  r[4]  = (__bf16)b.x; r[5]  = (__bf16)b.y; r[6]  = (__bf16)b.z; r[7]  = (__bf16)b.w;
  r[8]  = (__bf16)c.x; r[9]  = (__bf16)c.y; r[10] = (__bf16)c.z; r[11] = (__bf16)c.w;
  r[12] = (__bf16)d.x; r[13] = (__bf16)d.y; r[14] = (__bf16)d.z; r[15] = (__bf16)d.w;
  return r;
}

// A fragment (16x32 bf16, MxK): lane holds row M=lane&15, two contiguous
// 8-element K runs: [k0 + hi*8, +8) and [k0 + 16 + hi*8, +8).
__device__ __forceinline__ v16bf load_fragA(const float* __restrict__ row, int k0, int hi) {
  const float4* p0 = reinterpret_cast<const float4*>(row + k0 + hi * 8);
  const float4* p1 = reinterpret_cast<const float4*>(row + k0 + hi * 8 + 16);
  return pack16(p0[0], p0[1], p1[0], p1[1]);
}

// B fragment (32x16 bf16, KxN): lane holds column N=lane&15 as two contiguous
// 8-element K runs: [k0 + hi*8, +8) and [k0 + 16 + hi*8, +8) (== row of W since B = W^T).
__device__ __forceinline__ v16bf load_fragB(const float* __restrict__ row, int k0, int hi) {
  const float4* p0 = reinterpret_cast<const float4*>(row + k0 + hi * 8);
  const float4* p1 = reinterpret_cast<const float4*>(row + k0 + hi * 8 + 16);
  return pack16(p0[0], p0[1], p1[0], p1[1]);
}

// 16x16 output tile over full K=2048 using v_wmma_f32_16x16x32_bf16.
__device__ __forceinline__ v8f gemm_nt_tile(const float* __restrict__ Arow,
                                            const float* __restrict__ Brow,
                                            int hi) {
  v8f acc = {};
#pragma unroll 2
  for (int k0 = 0; k0 < D_DIM; k0 += 32) {
    v16bf af = load_fragA(Arow, k0, hi);
    v16bf bf = load_fragB(Brow, k0, hi);
    acc = __builtin_amdgcn_wmma_f32_16x16x32_bf16(false, af, false, bf,
                                                  (short)0, acc, false, false);
  }
  return acc;
}

// ---- kernel 1: fused QKV projections (6 GEMMs) ----------------------------

struct QkvArgs {
  const float* X1;            // [N1, D]
  const float* X2;            // [N2, D]
  const float* W[6];          // q1,k1,v1,q2,k2,v2 weights [D, D]
  const float* B[6];          // biases [D]
  float*       O[6];          // outputs: [N1,D] x3 then [N2,D] x3
};

// tiles: g<3: 16 mTiles x 128 nTiles = 2048 each; g>=3: 2 x 128 = 256 each.
// total waves = 3*2048 + 3*256 = 6912; 8 waves per 256-thread block.
__global__ __launch_bounds__(256) void qkv_kernel(QkvArgs a) {
  int wid  = (blockIdx.x * 256 + threadIdx.x) >> 5;
  int lane = threadIdx.x & 31;
  int hi   = lane >> 4;
  int ln   = lane & 15;

  int g, t;
  if (wid < 6144) { g = wid / 2048;       t = wid % 2048; }
  else            { g = 3 + (wid - 6144) / 256; t = (wid - 6144) % 256; }

  const int mTiles = (g < 3) ? 16 : 2;
  const int rows   = (g < 3) ? N1 : N2;
  const float* X   = (g < 3) ? a.X1 : a.X2;

  int nTile = t / mTiles;        // adjacent waves share W rows -> L2/L0 reuse
  int mTile = t % mTiles;

  int rowA = mTile * 16 + ln;  if (rowA >= rows) rowA = rows - 1;   // clamp loads
  int rowB = nTile * 16 + ln;                                       // always < 2048

  v8f acc = gemm_nt_tile(X + (size_t)rowA * D_DIM,
                         a.W[g] + (size_t)rowB * D_DIM, hi);

  int col    = nTile * 16 + ln;
  float bias = a.B[g][col];
  float* O   = a.O[g];
#pragma unroll
  for (int v = 0; v < 8; ++v) {
    int row = mTile * 16 + v + 8 * hi;
    if (row < rows) O[(size_t)row * D_DIM + col] = acc[v] + bias;
  }
}

// ---- kernel 2: sigmoid cross-attention scores -----------------------------

// waves 0..31:  probs1 = sigmoid(scale * q2 @ k1^T)  [27, 250]
// waves 32..63: probs2 = sigmoid(scale * q1 @ k2^T)  [250, 27]
__global__ __launch_bounds__(256) void probs_kernel(const float* __restrict__ q1,
                                                    const float* __restrict__ k1,
                                                    const float* __restrict__ q2,
                                                    const float* __restrict__ k2,
                                                    float* __restrict__ probs1,
                                                    float* __restrict__ probs2) {
  int wid  = (blockIdx.x * 256 + threadIdx.x) >> 5;
  int lane = threadIdx.x & 31;
  int hi   = lane >> 4;
  int ln   = lane & 15;

  const float scale = 0.022097086912079608f;   // 1/sqrt(2048)

  const float* A; const float* B; float* O;
  int rows, cols, mTiles;
  if (wid < 32) { A = q2; B = k1; O = probs1; rows = N2; cols = N1; mTiles = 2;  }
  else          { A = q1; B = k2; O = probs2; rows = N1; cols = N2; mTiles = 16; }
  int t = wid & 31;
  int nTile = t / mTiles;
  int mTile = t % mTiles;

  int rowA = mTile * 16 + ln;  if (rowA >= rows) rowA = rows - 1;
  int rowB = nTile * 16 + ln;  if (rowB >= cols) rowB = cols - 1;

  v8f acc = gemm_nt_tile(A + (size_t)rowA * D_DIM,
                         B + (size_t)rowB * D_DIM, hi);

  int col = nTile * 16 + ln;
  if (col < cols) {
#pragma unroll
    for (int v = 0; v < 8; ++v) {
      int row = mTile * 16 + v + 8 * hi;
      if (row < rows) {
        float x = acc[v] * scale;
        O[(size_t)row * cols + col] = 1.0f / (1.0f + __expf(-x));
      }
    }
  }
}

// ---- kernel 3: fc dot products (tiny) -------------------------------------

__global__ __launch_bounds__(256) void fc_kernel(const float* __restrict__ probs1,
                                                 const float* __restrict__ probs2,
                                                 const float* __restrict__ w_fc1,
                                                 const float* __restrict__ w_fc2,
                                                 float* __restrict__ s1,
                                                 float* __restrict__ s2) {
  int t = threadIdx.x;
  if (t < N2) {                       // s1[t] = probs1[t,:] . w_fc1   (len 250)
    float s = 0.0f;
    for (int i = 0; i < N1; ++i) s += probs1[t * N1 + i] * w_fc1[i];
    s1[t] = s;
  }
  if (t < N1) {                       // s2[t] = probs2[t,:] . w_fc2   (len 27)
    float s = 0.0f;
    for (int j = 0; j < N2; ++j) s += probs2[t * N2 + j] * w_fc2[j];
    s2[t] = s;
  }
}

// ---- kernel 4: context epilogue (memory-bound broadcast scale) ------------

__global__ __launch_bounds__(256) void ctx_kernel(const float* __restrict__ v1,
                                                  const float* __restrict__ v2,
                                                  const float* __restrict__ s1,
                                                  const float* __restrict__ s2,
                                                  const float* __restrict__ b_fc1,
                                                  const float* __restrict__ b_fc2,
                                                  float* __restrict__ ctx2,
                                                  float* __restrict__ ctx1) {
  int i = blockIdx.x * 256 + threadIdx.x;
  const int n2sz = N1 * D_DIM;            // 512000
  const int n1sz = N2 * D_DIM;            // 55296
  if (i < n2sz) {
    ctx2[i] = v1[i] * s2[i >> 11] + b_fc2[0];
  } else if (i < n2sz + n1sz) {
    int j = i - n2sz;
    ctx1[j] = v2[j] * s1[j >> 11] + b_fc1[0];
  }
}

// ---- launch ---------------------------------------------------------------

extern "C" void kernel_launch(void* const* d_in, const int* in_sizes, int n_in,
                              void* d_out, int out_size, void* d_ws, size_t ws_size,
                              hipStream_t stream) {
  // setup_inputs() order:
  // 0:x1 1:x2 2:Wq1 3:bq1 4:Wk1 5:bk1 6:Wv1 7:bv1
  // 8:Wq2 9:bq2 10:Wk2 11:bk2 12:Wv2 13:bv2 14:w_fc1 15:b_fc1 16:w_fc2 17:b_fc2
  const float* x1   = (const float*)d_in[0];
  const float* x2   = (const float*)d_in[1];
  const float* Wq1  = (const float*)d_in[2];  const float* bq1 = (const float*)d_in[3];
  const float* Wk1  = (const float*)d_in[4];  const float* bk1 = (const float*)d_in[5];
  const float* Wv1  = (const float*)d_in[6];  const float* bv1 = (const float*)d_in[7];
  const float* Wq2  = (const float*)d_in[8];  const float* bq2 = (const float*)d_in[9];
  const float* Wk2  = (const float*)d_in[10]; const float* bk2 = (const float*)d_in[11];
  const float* Wv2  = (const float*)d_in[12]; const float* bv2 = (const float*)d_in[13];
  const float* wfc1 = (const float*)d_in[14]; const float* bfc1 = (const float*)d_in[15];
  const float* wfc2 = (const float*)d_in[16]; const float* bfc2 = (const float*)d_in[17];

  // workspace layout (floats)
  float* ws = (float*)d_ws;
  float* q1 = ws;
  float* k1 = ws + 512000;
  float* v1 = ws + 1024000;
  float* q2 = ws + 1536000;
  float* k2 = ws + 1591296;
  float* v2 = ws + 1646592;
  float* s1 = ws + 1701888;
  float* s2 = ws + 1701920;

  // d_out layout (flat, return order): ctx2[250*2048], probs2[250*27],
  // ctx1[27*2048], probs1[27*250]
  float* out    = (float*)d_out;
  float* ctx2   = out;
  float* probs2 = out + 512000;
  float* ctx1   = out + 518750;
  float* probs1 = out + 574046;

  QkvArgs a;
  a.X1 = x1; a.X2 = x2;
  a.W[0] = Wq1; a.W[1] = Wk1; a.W[2] = Wv1; a.W[3] = Wq2; a.W[4] = Wk2; a.W[5] = Wv2;
  a.B[0] = bq1; a.B[1] = bk1; a.B[2] = bv1; a.B[3] = bq2; a.B[4] = bk2; a.B[5] = bv2;
  a.O[0] = q1;  a.O[1] = k1;  a.O[2] = v1;  a.O[3] = q2;  a.O[4] = k2;  a.O[5] = v2;

  // 6912 wave-tiles, 8 waves/block -> 864 blocks
  qkv_kernel<<<864, 256, 0, stream>>>(a);

  // 64 wave-tiles -> 8 blocks
  probs_kernel<<<8, 256, 0, stream>>>(q1, k1, q2, k2, probs1, probs2);

  fc_kernel<<<1, 256, 0, stream>>>(probs1, probs2, wfc1, wfc2, s1, s2);

  // 512000 + 55296 = 567296 elements
  ctx_kernel<<<(567296 + 255) / 256, 256, 0, stream>>>(v1, v2, s1, s2,
                                                       bfc1, bfc2, ctx2, ctx1);
}
